// DenoisingNetwork_39462159516118
// MI455X (gfx1250) — matrix operand order repaired and checked by reading, not compiled, not run on hardware
//
#include <hip/hip_runtime.h>
#include <math.h>

// ---------------------------------------------------------------------------
// Deformable-attention style network on MI455X (gfx1250, wave32, WMMA).
//
// Pipeline:
//   1) cvt:    value f32 -> bf16 (row major, A-operand of GEMM1)
//   2) pack:   W_v, W_out f32 -> bf16 in WMMA B-fragment layout (per-lane
//              contiguous 32B), so GEMM inner loop does no strided loads
//   3) gemm:   v = value @ W_v + b_v        (v_wmma_f32_16x16x32_bf16, f32 acc)
//   4) proj:   off / aw logits = query @ W_off|W_aw  (exact f32 -> indices)
//   5) sample: softmax(P=4) + bilinear gather over L, writes mid as bf16
//   6) gemm:   out = mid @ W_out + b_out    (v_wmma, f32 out)
//
// GEMM tiling: wave = 32x64 of C (2 A-frags x 4 B-frags -> 8 WMMA / K-step,
// 12 b128 loads / K-step). Block = 8 waves = 128x128 of C. Grid (N/128, M/128).
// ---------------------------------------------------------------------------

typedef __bf16 bf16;
typedef __attribute__((ext_vector_type(16))) __bf16 bf16x16;
typedef __attribute__((ext_vector_type(8)))  float  f32x8;

#define BATCH 8
#define LQ    2048
#define DIM   512
#define NH    8
#define NP    4
#define CH    64           // DIM / NH
#define MROWS (BATCH * LQ) // 16384

// ---------------------------------------------------------------------------
// f32 -> bf16 elementwise convert (row-major A operand)
// ---------------------------------------------------------------------------
__global__ void k_cvt_bf16(const float* __restrict__ x, bf16* __restrict__ y, int n) {
    int i = blockIdx.x * blockDim.x + threadIdx.x;
    if (i < n) y[i] = (bf16)x[i];
}

// ---------------------------------------------------------------------------
// Pack a row-major f32 weight matrix W (K x N) into per-lane WMMA B-fragment
// layout: Bp[kt][nt][lane][16 bf16], kt over K/32 tiles, nt over N/16 tiles.
// Lane layout (16x16x32 bf16 B operand, wave32):
//   n    = (lane & 15) + 16*nt
//   half = lane >> 4            -> K offset 16*half within the 32-K tile
//   VGPR j holds K = kbase+2j (lo16) and kbase+2j+1 (hi16)
// ---------------------------------------------------------------------------
__global__ void k_pack_b(const float* __restrict__ W, bf16* __restrict__ Bp,
                         int K, int N) {
    int t = blockIdx.x * blockDim.x + threadIdx.x;
    int ntile = N >> 4;
    int total = (K >> 5) * ntile * 32;
    if (t >= total) return;
    int lane = t & 31;
    int nt   = (t >> 5) % ntile;
    int kt   = (t >> 5) / ntile;
    int n     = nt * 16 + (lane & 15);
    int kbase = kt * 32 + ((lane >> 4) << 4);
    bf16* dst = Bp + (size_t)t * 16;
#pragma unroll
    for (int j = 0; j < 8; ++j) {
        dst[2 * j]     = (bf16)W[(size_t)(kbase + 2 * j)     * N + n];
        dst[2 * j + 1] = (bf16)W[(size_t)(kbase + 2 * j + 1) * N + n];
    }
}

// ---------------------------------------------------------------------------
// A fragment load: row-major bf16, lda = K. contiguous k order (same as packed B):
//   lane<16 : M=lane,    K in {0..15}
//   lane>=16: M=lane-16, K in {16..31}
// -> two contiguous 16-byte loads per lane.
// ---------------------------------------------------------------------------
__device__ __forceinline__ bf16x16 load_a_frag(const bf16* __restrict__ A,
                                               int lda, int lane) {
    int half = lane >> 4;
    int m    = lane & 15;
    const bf16* p = A + (size_t)m * lda + 16 * half;
    union { uint4 u[2]; bf16x16 v; } u;
    u.u[0] = *(const uint4*)(p);        // K = 16*half .. 16*half+7
    u.u[1] = *(const uint4*)(p + 8);    // K = 16*half+8 .. 16*half+15
    return u.v;
}

// ---------------------------------------------------------------------------
// bf16 GEMM: C(MxN f32) = A(MxK bf16 row-major) x Bp(packed frags) + bias(N)
// Block: 256 threads = 8 waves (4 M-waves x 2 N-waves) -> 128x128 C tile.
// Wave: 32 rows x 64 cols, 8 f32x8 accumulators, 8 WMMA per K-step.
// Grid: (N/128, M/128).
// ---------------------------------------------------------------------------
__global__ void k_gemm_bf16(const bf16* __restrict__ A,
                            const bf16x16* __restrict__ Bp,
                            const float* __restrict__ bias,
                            float* __restrict__ C,
                            int M, int N, int K) {
    const int wave  = threadIdx.x >> 5;
    const int lane  = threadIdx.x & 31;
    const int waveM = wave & 3;        // 0..3
    const int waveN = wave >> 2;       // 0..1
    const int m0 = blockIdx.y * 128 + waveM * 32;
    const int n0 = blockIdx.x * 128 + waveN * 64;
    const int ntiles = N >> 4;

    f32x8 acc[2][4] = {};

    for (int k0 = 0; k0 < K; k0 += 32) {
        const bf16x16 a0 = load_a_frag(A + (size_t)m0 * K + k0, K, lane);
        const bf16x16 a1 = load_a_frag(A + (size_t)(m0 + 16) * K + k0, K, lane);
        const size_t base = ((size_t)(k0 >> 5) * ntiles + (n0 >> 4)) * 32 + lane;
#pragma unroll
        for (int j = 0; j < 4; ++j) {
            const bf16x16 b = Bp[base + 32 * j];
            acc[0][j] = __builtin_amdgcn_wmma_f32_16x16x32_bf16(
                false, a0, false, b, (short)0, acc[0][j], false, false);
            acc[1][j] = __builtin_amdgcn_wmma_f32_16x16x32_bf16(
                false, a1, false, b, (short)0, acc[1][j], false, false);
        }
    }

    // C/D layout: lane<16 -> N=lane, VGPR r -> M=r ; lane>=16 -> M=r+8
    const int n  = lane & 15;
    const int mh = (lane >> 4) * 8;
#pragma unroll
    for (int mi = 0; mi < 2; ++mi) {
#pragma unroll
        for (int j = 0; j < 4; ++j) {
            const int nn = n0 + j * 16 + n;
            const float bv = bias[nn];
#pragma unroll
            for (int r = 0; r < 8; ++r) {
                C[(size_t)(m0 + 16 * mi + mh + r) * N + nn] = acc[mi][j][r] + bv;
            }
        }
    }
}

// ---------------------------------------------------------------------------
// Small projections in exact f32 (they determine gather indices):
//   off(row, 64) = q(row,512) @ W_off + b_off
//   awl(row, 32) = q(row,512) @ W_aw  + b_aw     (softmax applied later)
// One block per row; query row staged in LDS.
// ---------------------------------------------------------------------------
__global__ void k_proj(const float* __restrict__ q,
                       const float* __restrict__ W_off, const float* __restrict__ b_off,
                       const float* __restrict__ W_aw,  const float* __restrict__ b_aw,
                       float* __restrict__ off, float* __restrict__ awl) {
    __shared__ float qs[DIM];
    const int row = blockIdx.x;
    for (int i = threadIdx.x; i < DIM; i += blockDim.x)
        qs[i] = q[(size_t)row * DIM + i];
    __syncthreads();

    const int c = threadIdx.x;
    if (c < 64) {
        float acc = b_off[c];
        for (int k = 0; k < DIM; ++k) acc = fmaf(qs[k], W_off[(size_t)k * 64 + c], acc);
        off[(size_t)row * 64 + c] = acc;
    } else if (c < 96) {
        const int c2 = c - 64;
        float acc = b_aw[c2];
        for (int k = 0; k < DIM; ++k) acc = fmaf(qs[k], W_aw[(size_t)k * 32 + c2], acc);
        awl[(size_t)row * 32 + c2] = acc;
    }
}

// ---------------------------------------------------------------------------
// Sampling: per (b,l,h) one wave32. softmax over P=4, bilinear gather of v
// rows (L2-resident), weighted sum; writes mid as bf16 (A-operand of GEMM2).
// Each lane covers channels c = lane and lane+32 (C = 64).
// ---------------------------------------------------------------------------
__global__ void k_sample(const float* __restrict__ v,
                         const float* __restrict__ off,
                         const float* __restrict__ awl,
                         bf16* __restrict__ mid) {
    const int gw   = blockIdx.x * (blockDim.x >> 5) + (threadIdx.x >> 5);
    const int lane = threadIdx.x & 31;
    if (gw >= MROWS * NH) return;

    const int h  = gw % NH;
    const int bl = gw / NH;          // b*L + l
    const int l  = bl % LQ;
    const int b  = bl / LQ;
    const float refy = (float)l * (1.0f / (float)(LQ - 1));

    const float* offp = off + (size_t)bl * (NH * NP * 2) + h * (NP * 2);
    const float* awp  = awl + (size_t)bl * (NH * NP) + h * NP;

    // softmax over P=4 (redundant per-lane scalar work; cache-broadcast)
    float a0 = awp[0], a1 = awp[1], a2 = awp[2], a3 = awp[3];
    float mx = fmaxf(fmaxf(a0, a1), fmaxf(a2, a3));
    float w0 = __expf(a0 - mx), w1 = __expf(a1 - mx);
    float w2 = __expf(a2 - mx), w3 = __expf(a3 - mx);
    const float inv = 1.0f / (w0 + w1 + w2 + w3);
    float wgt[NP] = {w0 * inv, w1 * inv, w2 * inv, w3 * inv};

    const float* vb = v + (size_t)b * LQ * DIM + h * CH;
    float acc0 = 0.0f, acc1 = 0.0f;

#pragma unroll
    for (int p = 0; p < NP; ++p) {
        const float ox = offp[2 * p];
        const float oy = offp[2 * p + 1];
        const float lx = fminf(fmaxf(ox, 0.0f), 1.0f);
        const float ly = fminf(fmaxf(oy + refy, 0.0f), 1.0f);
        const float ix = ((lx + 1.0f) * (float)LQ - 1.0f) * 0.5f;
        const float iy = ly * 0.5f;
        const float ix0 = floorf(ix);
        const float fx  = ix - ix0;
        const float iy0 = floorf(iy);
        const float fy  = iy - iy0;
        const float ywt = (1.0f - fy) * (iy0 == 0.0f ? 1.0f : 0.0f)
                        + fy * ((iy0 + 1.0f) == 0.0f ? 1.0f : 0.0f);
        const int i0 = (int)ix0;
        const int i1 = i0 + 1;
        const float m0 = (i0 >= 0 && i0 < LQ) ? 1.0f : 0.0f;
        const float m1 = (i1 >= 0 && i1 < LQ) ? 1.0f : 0.0f;
        const int i0c = min(max(i0, 0), LQ - 1);
        const int i1c = min(max(i1, 0), LQ - 1);

        const float* r0 = vb + (size_t)i0c * DIM;
        const float* r1 = vb + (size_t)i1c * DIM;
        const float g0a = r0[lane]      * m0;
        const float g0b = r0[lane + 32] * m0;
        const float g1a = r1[lane]      * m1;
        const float g1b = r1[lane + 32] * m1;

        const float sa = ywt * ((1.0f - fx) * g0a + fx * g1a);
        const float sb = ywt * ((1.0f - fx) * g0b + fx * g1b);
        acc0 = fmaf(wgt[p], sa, acc0);
        acc1 = fmaf(wgt[p], sb, acc1);
    }

    bf16* dst = mid + (size_t)bl * DIM + h * CH;
    dst[lane]      = (bf16)acc0;
    dst[lane + 32] = (bf16)acc1;
}

// ---------------------------------------------------------------------------
// Host-side launcher
// ---------------------------------------------------------------------------
extern "C" void kernel_launch(void* const* d_in, const int* in_sizes, int n_in,
                              void* d_out, int out_size, void* d_ws, size_t ws_size,
                              hipStream_t stream) {
    (void)in_sizes; (void)n_in; (void)out_size; (void)ws_size;

    const float* query = (const float*)d_in[0];
    /* d_in[1] = key_in : unused by reference */
    const float* value = (const float*)d_in[2];
    const float* W_v   = (const float*)d_in[3];
    const float* b_v   = (const float*)d_in[4];
    const float* W_off = (const float*)d_in[5];
    const float* b_off = (const float*)d_in[6];
    const float* W_aw  = (const float*)d_in[7];
    const float* b_aw  = (const float*)d_in[8];
    const float* W_out = (const float*)d_in[9];
    const float* b_out = (const float*)d_in[10];
    float* out = (float*)d_out;

    // workspace carve-up (256B aligned)
    char* ws = (char*)d_ws;
    size_t o = 0;
    auto carve = [&](size_t bytes) -> void* {
        void* p = ws + o;
        o = (o + bytes + 255) & ~(size_t)255;
        return p;
    };
    bf16*  value_bf = (bf16*) carve((size_t)MROWS * DIM * sizeof(bf16));
    bf16*  mid_bf   = (bf16*) carve((size_t)MROWS * DIM * sizeof(bf16));
    bf16*  Wv_p     = (bf16*) carve((size_t)DIM * DIM * sizeof(bf16));
    bf16*  Wout_p   = (bf16*) carve((size_t)DIM * DIM * sizeof(bf16));
    float* v_f32    = (float*)carve((size_t)MROWS * DIM * sizeof(float));
    float* off_f    = (float*)carve((size_t)MROWS * 64 * sizeof(float));
    float* awl_f    = (float*)carve((size_t)MROWS * 32 * sizeof(float));

    // 1) value -> bf16
    {
        int n = MROWS * DIM;
        k_cvt_bf16<<<(n + 255) / 256, 256, 0, stream>>>(value, value_bf, n);
    }
    // 2) pack weights into B-fragment layout
    {
        int total = (DIM >> 5) * (DIM >> 4) * 32;   // 16384
        k_pack_b<<<(total + 255) / 256, 256, 0, stream>>>(W_v,   Wv_p,   DIM, DIM);
        k_pack_b<<<(total + 255) / 256, 256, 0, stream>>>(W_out, Wout_p, DIM, DIM);
    }
    // 3) v = value @ W_v + b_v   (WMMA bf16, f32 out)
    {
        dim3 grid(DIM / 128, MROWS / 128);
        k_gemm_bf16<<<grid, 256, 0, stream>>>(value_bf, (const bf16x16*)Wv_p,
                                              b_v, v_f32, MROWS, DIM, DIM);
    }
    // 4) offsets / attention-weight logits (exact f32)
    k_proj<<<MROWS, 128, 0, stream>>>(query, W_off, b_off, W_aw, b_aw, off_f, awl_f);

    // 5) softmax + bilinear gather-weighted sum -> mid (bf16)
    {
        int waves = MROWS * NH;                      // 131072
        k_sample<<<waves / 8, 256, 0, stream>>>(v_f32, off_f, awl_f, mid_bf);
    }
    // 6) out = mid @ W_out + b_out  (WMMA bf16, f32 out)
    {
        dim3 grid(DIM / 128, MROWS / 128);
        k_gemm_bf16<<<grid, 256, 0, stream>>>(mid_bf, (const bf16x16*)Wout_p,
                                              b_out, out, MROWS, DIM, DIM);
    }
}
